// DGST_12567074308986
// MI455X (gfx1250) — compile-verified, matrix operand order checked by reading, not run on hardware
//
#include <hip/hip_runtime.h>
#include <hip/hip_bf16.h>

// ---------------- problem constants ----------------
constexpr int kL = 4;        // layers
constexpr int kB = 32;       // batch
constexpr int kH = 128;      // hidden
constexpr int kV = 28996;    // vocab
constexpr int kS = 128;      // seq len
constexpr int kG = 4 * kH;   // 512 gate width
constexpr int kX = 2 * kH;   // 256 concat width

typedef __bf16 bf16;
typedef __attribute__((ext_vector_type(16))) __bf16 bf16x16;
typedef __attribute__((ext_vector_type(8)))  __bf16 bf16x8;
typedef __attribute__((ext_vector_type(8)))  float  f32x8;

// TDM descriptor vector types (6-arg builtin signature: clang-23 toolchain)
typedef __attribute__((ext_vector_type(4))) unsigned int uint32x4;
typedef __attribute__((ext_vector_type(8))) int          int32x8;
typedef __attribute__((ext_vector_type(4))) int          int32x4;

// ---------------- workspace layout (bytes) ----------------
constexpr size_t WIH0_OFF  = 0;                                             // 2*512*128 bf16
constexpr size_t WREST_OFF = WIH0_OFF  + sizeof(bf16) * 2 * kG * kH;        // 3*2*512*256
constexpr size_t WHH_OFF   = WREST_OFF + sizeof(bf16) * 3 * 2 * kG * kX;    // 4*2*512*128
constexpr size_t WD_OFF    = WHH_OFF   + sizeof(bf16) * kL * 2 * kG * kH;   // V*256
constexpr size_t EMB_OFF   = WD_OFF    + sizeof(bf16) * (size_t)kV * kX;    // V*128
constexpr size_t CST_OFF   = EMB_OFF   + sizeof(bf16) * (size_t)kV * kH;    // f32 L*2*B*H
constexpr size_t HST_OFF   = CST_OFF   + sizeof(float) * kL * 2 * kB * kH;  // bf16 L*2*B*H
constexpr size_t X0_OFF    = HST_OFF   + sizeof(bf16) * kL * 2 * kB * kH;   // bf16 B*X
constexpr size_t XF_OFF    = X0_OFF    + sizeof(bf16) * kB * kX;            // bf16 B*X
constexpr size_t GATE_OFF  = XF_OFF    + sizeof(bf16) * kB * kX;            // f32 2*B*G

// ---------------- Tensor Data Mover: 2D bf16 tile descriptor ----------------
// D# per ISA 08_async_tensor §8.3/8.4. 2D tensor: groups 2/3 zero.
// dim0/dim1/stride0 in elements (data_size = 2 bytes).
__device__ __forceinline__ void tdm_desc_2d_bf16(const void* gptr, unsigned lds_addr,
                                                 int dim0, int dim1, int stride0,
                                                 uint32x4* g0o, int32x8* g1o) {
  unsigned long long ga = (unsigned long long)gptr;
  uint32x4 g0;
  g0.x = 1u;                                        // count=1 (valid user D#)
  g0.y = lds_addr;                                  // LDS byte address
  g0.z = (unsigned)(ga & 0xffffffffu);              // global_addr[31:0]
  g0.w = (unsigned)((ga >> 32) & 0x01ffffffu)       // global_addr[56:32]
       | (2u << 30);                                // type=2 ("image")
  int32x8 g1;
  g1[0] = 1 << 16;                                  // data_size=1 (2 bytes)
  g1[1] = (dim0 & 0xffff) << 16;                    // tensor_dim0[15:0]
  g1[2] = ((dim0 >> 16) & 0xffff)                   // tensor_dim0[31:16]
        | ((dim1 & 0xffff) << 16);                  // tensor_dim1[15:0]
  g1[3] = ((dim1 >> 16) & 0xffff)                   // tensor_dim1[31:16]
        | ((dim0 & 0xffff) << 16);                  // tile_dim0 = dim0
  g1[4] = dim1 & 0xffff;                            // tile_dim1 = dim1 (tile_dim2=0)
  g1[5] = stride0;                                  // tensor_dim0_stride[31:0]
  g1[6] = 0;                                        // stride0[47:32] | dim1_stride lo
  g1[7] = 0;
  *g0o = g0;
  *g1o = g1;
}

__device__ __forceinline__ void tdm_load_2d_bf16(const void* gptr, unsigned lds_addr,
                                                 int dim0, int dim1, int stride0) {
  uint32x4 g0; int32x8 g1;
  tdm_desc_2d_bf16(gptr, lds_addr, dim0, dim1, stride0, &g0, &g1);
  int32x4 z4 = {0, 0, 0, 0};
  int32x8 z8 = {0, 0, 0, 0, 0, 0, 0, 0};
  __builtin_amdgcn_tensor_load_to_lds(g0, g1, z4, z4, z8, 0);
}

__device__ __forceinline__ void tdm_store_2d_bf16(void* gptr, unsigned lds_addr,
                                                  int dim0, int dim1, int stride0) {
  uint32x4 g0; int32x8 g1;
  tdm_desc_2d_bf16(gptr, lds_addr, dim0, dim1, stride0, &g0, &g1);
  int32x4 z4 = {0, 0, 0, 0};
  int32x8 z8 = {0, 0, 0, 0, 0, 0, 0, 0};
  __builtin_amdgcn_tensor_store_from_lds(g0, g1, z4, z4, z8, 0);
}

// ---------------- WMMA helpers (ISA 7.12.2 16-bit layouts) ----------------
// A/B operand: lane r=lane&15 selects row (A: M, B: N); lane half selects the
// K-subgroup: half==0 -> K {k0+0..7, k0+16..23}, half==1 -> K {k0+8..15, k0+24..31}.
__device__ __forceinline__ bf16x16 load_tile16(const bf16* base, int ld, int r0,
                                               int k0, int lane) {
  int half = lane >> 4;
  int r    = lane & 15;
  const bf16* p = base + (size_t)(r0 + r) * ld + k0 + half * 8;
  bf16x16 v;
  *reinterpret_cast<bf16x8*>(&v)       = *reinterpret_cast<const bf16x8*>(p);
  *(reinterpret_cast<bf16x8*>(&v) + 1) = *reinterpret_cast<const bf16x8*>(p + 16);
  return v;
}

__device__ __forceinline__ f32x8 wmma_bf16(bf16x16 a, bf16x16 b, f32x8 c) {
  return __builtin_amdgcn_wmma_f32_16x16x32_bf16(
      /*neg_a=*/false, a, /*neg_b=*/false, b,
      /*c_mod=*/(short)0, c, /*reuse_a=*/false, /*reuse_b=*/false);
}

// ---------------- f32 -> bf16 conversion ----------------
__global__ void cvt_bf16_kernel(const float* __restrict__ src,
                                bf16* __restrict__ dst, int n) {
  int i = blockIdx.x * blockDim.x + threadIdx.x;
  if (i < n) dst[i] = (bf16)src[i];
}

// ---------------- state init + first-token gather ----------------
__global__ void init_kernel(const int* __restrict__ data,
                            const bf16* __restrict__ embb,
                            bf16* __restrict__ x0,
                            float* __restrict__ c_state,
                            bf16* __restrict__ h_state) {
  int tid = blockIdx.x * blockDim.x + threadIdx.x;
  if (tid < kL * 2 * kB * kH) {
    c_state[tid] = 0.0f;
    h_state[tid] = (bf16)0.0f;
  }
  if (tid < kB * kH) {
    int b = tid / kH, j = tid % kH;
    int idx = data[b * kS + 0];
    x0[b * kX + j] = embb[(size_t)idx * kH + j];
  }
}

// ---------------- one LSTM time-step: all 4 layers x 2 dirs ----------------
// Single workgroup (32 waves). Per layer: 2dir*2Mtile*32Ntile = 128 WMMA tiles,
// 4 tiles per wave. x kept in LDS across layers; gates through L2 scratch.
__global__ __launch_bounds__(1024)
void lstm_step_kernel(const bf16* __restrict__ wih0,   // (2,512,128)
                      const bf16* __restrict__ wrest,  // (3,2,512,256)
                      const bf16* __restrict__ whh,    // (4,2,512,128)
                      const float* __restrict__ b_ih,  // (4,2,512) f32
                      const float* __restrict__ b_hh,  // (4,2,512) f32
                      const bf16* __restrict__ x0,     // (B,X) cols 0..H-1 valid
                      bf16* __restrict__ xf,           // out (B,X)
                      float* __restrict__ c_state,     // (L,2,B,H) f32
                      bf16* __restrict__ h_state,      // (L,2,B,H) bf16
                      float* __restrict__ gates) {     // scratch (2,B,G) f32
  __shared__ __align__(16) bf16 x_lds[kB * kX];
  const int tid  = threadIdx.x;
  const int lane = tid & 31;
  const int wave = tid >> 5;

  // TDM-stage step input into LDS (32x256 bf16 tile), wave 0 drives the DMA
  if (wave == 0) {
    tdm_load_2d_bf16(x0, (unsigned)(size_t)&x_lds[0], kX, kB, kX);
    __builtin_amdgcn_s_wait_tensorcnt(0);
  }
  __syncthreads();

  for (int l = 0; l < kL; ++l) {
    const int Kx = (l == 0) ? kH : kX;

    // ---- gate GEMMs: gates = x @ Wi^T + h @ Whh^T + bias ----
    #pragma unroll
    for (int i = 0; i < 4; ++i) {
      int t   = wave * 4 + i;        // 0..127
      int dir = t >> 6;
      int rem = t & 63;
      int m0  = (rem >> 5) * 16;     // 0 or 16
      int n0  = (rem & 31) * 16;     // 0..496
      const bf16* Wi = (l == 0) ? (wih0 + (size_t)dir * kG * kH)
                                : (wrest + (size_t)((l - 1) * 2 + dir) * kG * kX);
      const bf16* Wh = whh + (size_t)(l * 2 + dir) * kG * kH;
      const bf16* hp = h_state + (size_t)(l * 2 + dir) * kB * kH;

      f32x8 acc = {};
      for (int k0 = 0; k0 < Kx; k0 += 32) {
        bf16x16 a = load_tile16(x_lds, kX, m0, k0, lane);
        bf16x16 w = load_tile16(Wi, Kx, n0, k0, lane);
        acc = wmma_bf16(a, w, acc);
      }
      #pragma unroll
      for (int k0 = 0; k0 < kH; k0 += 32) {
        bf16x16 a = load_tile16(hp, kH, m0, k0, lane);
        bf16x16 w = load_tile16(Wh, kH, n0, k0, lane);
        acc = wmma_bf16(a, w, acc);
      }
      // C layout: n = n0 + (lane&15); m = m0 + vgpr + 8*(lane>=16)
      int n     = n0 + (lane & 15);
      int mbase = m0 + ((lane >> 4) << 3);
      float bias = b_ih[(l * 2 + dir) * kG + n] + b_hh[(l * 2 + dir) * kG + n];
      float* gout = gates + (size_t)dir * kB * kG;
      #pragma unroll
      for (int r = 0; r < 8; ++r)
        gout[(size_t)(mbase + r) * kG + n] = acc[r] + bias;
    }
    __syncthreads();

    // ---- pointwise LSTM update: 2*B*H = 8192 elements ----
    for (int e = tid; e < 2 * kB * kH; e += 1024) {
      int dir = e >> 12;          // B*H == 4096
      int rem = e & 4095;
      int m   = rem >> 7;
      int j   = rem & 127;
      const float* g = gates + (size_t)dir * kB * kG + (size_t)m * kG;
      float gi = g[j];
      float gf = g[kH + j];
      float gg = g[2 * kH + j];
      float go = g[3 * kH + j];
      gi = 1.0f / (1.0f + __expf(-gi));
      gf = 1.0f / (1.0f + __expf(-gf));
      gg = tanhf(gg);
      go = 1.0f / (1.0f + __expf(-go));
      float* cp = c_state + (size_t)(l * 2 + dir) * kB * kH + (size_t)m * kH + j;
      float c2 = gf * (*cp) + gi * gg;
      float h2 = go * tanhf(c2);
      *cp = c2;
      bf16 hb = (bf16)h2;
      h_state[(size_t)(l * 2 + dir) * kB * kH + (size_t)m * kH + j] = hb;
      x_lds[m * kX + dir * kH + j] = hb;   // next layer input: concat(hf, hb)
    }
    __syncthreads();
  }

  // publish final x via TDM store (S_ENDPGM's implicit wait-idle covers TENSORcnt)
  if (wave == 0) {
    tdm_store_2d_bf16(xf, (unsigned)(size_t)&x_lds[0], kX, kB, kX);
    __builtin_amdgcn_s_wait_tensorcnt(0);
  }
}

// ---------------- logits = x @ W_dense^T ----------------
constexpr int kNB     = 128;                       // N columns per block
constexpr int kBlocks = (kV + kNB - 1) / kNB;      // 227

__global__ __launch_bounds__(256)
void dense_kernel(const bf16* __restrict__ x,    // (B,X)
                  const bf16* __restrict__ wd,   // (V,X) row-major, K contiguous
                  float* __restrict__ out,       // (B,S,V)
                  int s) {
  __shared__ __align__(16) bf16 x_lds[kB * kX];
  const int tid  = threadIdx.x;
  const int lane = tid & 31;
  const int wave = tid >> 5;

  // TDM-stage the shared A matrix while waves set up B addressing
  if (wave == 0) {
    tdm_load_2d_bf16(x, (unsigned)(size_t)&x_lds[0], kX, kB, kX);
    __builtin_amdgcn_s_wait_tensorcnt(0);
  }
  __syncthreads();

  const int n0 = blockIdx.x * kNB + wave * 16;
  f32x8 acc0 = {}, acc1 = {};
  #pragma unroll
  for (int k0 = 0; k0 < kX; k0 += 32) {
    // B operand straight from L2-resident W_dense (rows are K-contiguous)
    bf16x16 wtile;
    {
      int half = lane >> 4, r = lane & 15;
      int n = n0 + r;
      if (n >= kV) n = kV - 1;               // clamp edge tile
      const bf16* p = wd + (size_t)n * kX + k0 + half * 8;
      __builtin_prefetch(p + 32, 0, 0);      // global_prefetch_b8, next k-tile
      *reinterpret_cast<bf16x8*>(&wtile)       = *reinterpret_cast<const bf16x8*>(p);
      *(reinterpret_cast<bf16x8*>(&wtile) + 1) = *reinterpret_cast<const bf16x8*>(p + 16);
    }
    bf16x16 a0 = load_tile16(x_lds, kX, 0,  k0, lane);
    bf16x16 a1 = load_tile16(x_lds, kX, 16, k0, lane);
    acc0 = wmma_bf16(a0, wtile, acc0);
    acc1 = wmma_bf16(a1, wtile, acc1);
  }

  int n = n0 + (lane & 15);
  if (n < kV) {
    int mb = (lane >> 4) << 3;
    #pragma unroll
    for (int r = 0; r < 8; ++r) {
      out[(size_t)(mb + r) * kS * kV + (size_t)s * kV + n]      = acc0[r];
      out[(size_t)(mb + r + 16) * kS * kV + (size_t)s * kV + n] = acc1[r];
    }
  }
}

// ---------------- per-row argmax over V + embedding gather ----------------
__global__ __launch_bounds__(256)
void argmax_gather_kernel(const float* __restrict__ out,  // (B,S,V)
                          const bf16* __restrict__ embb,  // (V,H)
                          bf16* __restrict__ x0,          // (B,X)
                          int s) {
  __shared__ float svals[256];
  __shared__ int   sidx[256];
  const int b   = blockIdx.x;
  const int tid = threadIdx.x;
  const float* row = out + (size_t)b * kS * kV + (size_t)s * kV;

  float best = -3.402823466e38f;
  int   bi   = 0;
  for (int n = tid; n < kV; n += 256) {
    float v = row[n];
    if (v > best) { best = v; bi = n; }
  }
  svals[tid] = best;
  sidx[tid]  = bi;
  __syncthreads();
  for (int off = 128; off > 0; off >>= 1) {
    if (tid < off) {
      float v2 = svals[tid + off];
      int   i2 = sidx[tid + off];
      if (v2 > svals[tid] || (v2 == svals[tid] && i2 < sidx[tid])) {
        svals[tid] = v2;
        sidx[tid]  = i2;
      }
    }
    __syncthreads();
  }
  int idx = sidx[0];
  if (tid < kH) x0[b * kX + tid] = embb[(size_t)idx * kH + tid];
}

// ---------------- host driver ----------------
extern "C" void kernel_launch(void* const* d_in, const int* in_sizes, int n_in,
                              void* d_out, int out_size, void* d_ws, size_t ws_size,
                              hipStream_t stream) {
  const int*   data  = (const int*)d_in[0];
  const float* emb   = (const float*)d_in[1];
  const float* wih0  = (const float*)d_in[2];
  const float* wrest = (const float*)d_in[3];
  const float* whh   = (const float*)d_in[4];
  const float* bih   = (const float*)d_in[5];
  const float* bhh   = (const float*)d_in[6];
  const float* wd    = (const float*)d_in[7];
  float* out = (float*)d_out;

  char* ws = (char*)d_ws;
  bf16*  wih0_b  = (bf16*)(ws + WIH0_OFF);
  bf16*  wrest_b = (bf16*)(ws + WREST_OFF);
  bf16*  whh_b   = (bf16*)(ws + WHH_OFF);
  bf16*  wd_b    = (bf16*)(ws + WD_OFF);
  bf16*  emb_b   = (bf16*)(ws + EMB_OFF);
  float* cst     = (float*)(ws + CST_OFF);
  bf16*  hst     = (bf16*)(ws + HST_OFF);
  bf16*  x0      = (bf16*)(ws + X0_OFF);
  bf16*  xf      = (bf16*)(ws + XF_OFF);
  float* gates   = (float*)(ws + GATE_OFF);

  auto cvt = [&](const float* src, bf16* dst, int n) {
    cvt_bf16_kernel<<<(n + 255) / 256, 256, 0, stream>>>(src, dst, n);
  };
  cvt(wih0,  wih0_b,  2 * kG * kH);
  cvt(wrest, wrest_b, 3 * 2 * kG * kX);
  cvt(whh,   whh_b,   kL * 2 * kG * kH);
  cvt(wd,    wd_b,    kV * kX);
  cvt(emb,   emb_b,   kV * kH);

  init_kernel<<<(kL * 2 * kB * kH + 255) / 256, 256, 0, stream>>>(
      data, emb_b, x0, cst, hst);

  for (int s = 0; s < kS; ++s) {
    lstm_step_kernel<<<1, 1024, 0, stream>>>(
        wih0_b, wrest_b, whh_b, bih, bhh, x0, xf, cst, hst, gates);
    dense_kernel<<<kBlocks, 256, 0, stream>>>(xf, wd_b, out, s);
    argmax_gather_kernel<<<kB, 256, 0, stream>>>(out, emb_b, x0, s);
  }
}
